// DSGIAT_GraphBranch_15831249453409
// MI455X (gfx1250) — matrix operand order repaired and checked by reading, not compiled, not run on hardware
//
#include <hip/hip_runtime.h>

#define HID 128
#define HEADS 4
#define OUT1 512
#define IN_CH 256
#define JK 1280
#define NGRAPH 64
#define MLP_H 256
#define MLP_O 128
#define NEG_SLOPE 0.2f
#define EPS_DEN 1e-16f

typedef __attribute__((ext_vector_type(16))) __bf16 v16bf;
typedef __attribute__((ext_vector_type(8)))  float v8f;

// ---------------------------------------------------------------- utilities

__device__ inline float atomicMaxF(float* addr, float v) {
  // monotone bit-pun trick: signed max for non-negative, unsigned min for negative
  if (v >= 0.0f)
    return __int_as_float(atomicMax((int*)addr, __float_as_int(v)));
  else
    return __uint_as_float(atomicMin((unsigned int*)addr, __float_as_uint(v)));
}

__device__ inline unsigned short f2bf(float f) {
  unsigned int u = __float_as_uint(f);
  u += 0x7FFFu + ((u >> 16) & 1u);   // round-to-nearest-even
  return (unsigned short)(u >> 16);
}

__global__ void k_fill(float* __restrict__ p, float v, long n) {
  long i = (long)blockIdx.x * blockDim.x + threadIdx.x;
  long st = (long)gridDim.x * blockDim.x;
  for (; i < n; i += st) p[i] = v;
}

__global__ void k_f32_to_bf16(const float* __restrict__ in,
                              unsigned short* __restrict__ out, long n) {
  long i = (long)blockIdx.x * blockDim.x + threadIdx.x;
  long st = (long)gridDim.x * blockDim.x;
  for (; i < n; i += st) out[i] = f2bf(in[i]);
}

// ---------------------------------------------------------------- graph prep

__global__ void k_deg(const int* __restrict__ dst, float* __restrict__ deg, int E) {
  int e = blockIdx.x * blockDim.x + threadIdx.x;
  if (e < E) atomicAdd(&deg[dst[e]], 1.0f);
}

__global__ void k_edgew(const int* __restrict__ src, const int* __restrict__ dst,
                        const float* __restrict__ deg, float* __restrict__ ew, int E) {
  int e = blockIdx.x * blockDim.x + threadIdx.x;
  if (e >= E) return;
  float ds_ = deg[src[e]], dd = deg[dst[e]];
  float a = (ds_ > 0.0f) ? rsqrtf(ds_) : 0.0f;
  float b = (dd  > 0.0f) ? rsqrtf(dd)  : 0.0f;
  ew[e] = a * b;
}

// ---------------------------------------------------------------- WMMA GEMM
// C[M,Nc] = A[M,K] * B[K,Nc], A/B bf16 row-major, C f32. One 16x16 tile per wave.

__global__ void k_gemm_bf16(const __bf16* __restrict__ A, const __bf16* __restrict__ B,
                            float* __restrict__ C, int M, int Nc, int K,
                            int tilesN, int totalTiles) {
  int wid  = blockIdx.x * (blockDim.x >> 5) + (threadIdx.x >> 5);
  int lane = threadIdx.x & 31;
  if (wid >= totalTiles) return;
  int bm = wid / tilesN;
  int bn = wid % tilesN;

  int half = lane >> 4;            // lane half selects K sub-range
  int l15  = lane & 15;
  int row  = bm * 16 + l15;        // A row for this lane (both halves)
  int rowc = row < M ? row : M - 1;

  const __bf16* arow = A + (long)rowc * K;
  v8f acc = {};

  for (int k0 = 0; k0 < K; k0 += 32) {
    // A fragment: 16-bit A 16x32 layout — lane l15 = row; VGPR0-3 hold
    // K = half*8 + [0..7], VGPR4-7 hold K = 16 + half*8 + [0..7]
    union { uint4 q[2]; v16bf v; } ua;
    const __bf16* p0 = arow + k0 + half * 8;
    ua.q[0] = *(const uint4*)(p0);
    ua.q[1] = *(const uint4*)(p0 + 16);

    // B fragment: 16-bit B 32x16 layout — lane holds column N = l15,
    // K = half*8 + [0..7] then K = 16 + half*8 + [0..7]
    union { uint4 q[2]; v16bf v; } ub;
    const __bf16* pb = B + (long)(k0 + half * 8) * Nc + bn * 16 + l15;
    for (int s = 0; s < 8; s++) {
      ub.v[s] = pb[(long)s * Nc];
      ub.v[s + 8] = pb[(long)(s + 16) * Nc];
    }

    acc = __builtin_amdgcn_wmma_f32_16x16x32_bf16(
        /*neg_a=*/false, ua.v, /*neg_b=*/false, ub.v,
        /*c_mod=*/(short)0, acc, /*reuse_a=*/false, /*reuse_b=*/false);
  }

  // C/D 32-bit layout: VGPR r -> M = r + 8*half, N = l15
  int col = bn * 16 + l15;
#pragma unroll
  for (int r = 0; r < 8; r++) {
    int crow = bm * 16 + r + 8 * half;
    if (crow < M) C[(long)crow * Nc + col] = acc[r];
  }
}

// ---------------------------------------------------------------- GAT attention

// e_s[n,h] = <h[n,h,:], a_src[h,:]>, e_d likewise. One wave per (node, head).
__global__ void k_attn_logits(const float* __restrict__ h,
                              const float* __restrict__ a_src,
                              const float* __restrict__ a_dst,
                              float* __restrict__ es, float* __restrict__ ed, int N) {
  int wid  = blockIdx.x * (blockDim.x >> 5) + (threadIdx.x >> 5);
  int lane = threadIdx.x & 31;
  int n  = wid >> 2;
  int hd = wid & 3;
  if (n >= N) return;
  const float* hp = h + (long)n * OUT1 + hd * HID;
  const float* as = a_src + hd * HID;
  const float* ad = a_dst + hd * HID;
  float s1 = 0.0f, s2 = 0.0f;
  for (int c = lane; c < HID; c += 32) {
    float hv = hp[c];
    s1 += hv * as[c];
    s2 += hv * ad[c];
  }
#pragma unroll
  for (int o = 16; o > 0; o >>= 1) {
    s1 += __shfl_xor(s1, o, 32);
    s2 += __shfl_xor(s2, o, 32);
  }
  if (lane == 0) { es[n * HEADS + hd] = s1; ed[n * HEADS + hd] = s2; }
}

// per conv-edge (incl. self-loops): logits + segment max. One thread per edge.
__global__ void k_edge_logits_max(const int* __restrict__ src, const int* __restrict__ dst,
                                  const float* __restrict__ es, const float* __restrict__ ed,
                                  float* __restrict__ eedge, float* __restrict__ mmax,
                                  int E, int CE) {
  int e = blockIdx.x * blockDim.x + threadIdx.x;
  if (e >= CE) return;
  int s, d;
  if (e < E) { s = src[e]; d = dst[e]; } else { s = d = e - E; }
#pragma unroll
  for (int hd = 0; hd < HEADS; hd++) {
    float v = es[s * HEADS + hd] + ed[d * HEADS + hd];
    v = (v > 0.0f) ? v : NEG_SLOPE * v;
    eedge[(long)e * HEADS + hd] = v;
    atomicMaxF(&mmax[d * HEADS + hd], v);
  }
}

__global__ void k_fix_max(float* __restrict__ m, long n) {
  long i = (long)blockIdx.x * blockDim.x + threadIdx.x;
  if (i < n && !isfinite(m[i])) m[i] = 0.0f;
}

// exp + denom + weighted message scatter. One wave per (conv-edge, head).
__global__ void k_edge_aggregate(const int* __restrict__ src, const int* __restrict__ dst,
                                 const float* __restrict__ eedge, const float* __restrict__ mmax,
                                 float* __restrict__ denom,
                                 const float* __restrict__ h, float* __restrict__ acc,
                                 int E, int CE) {
  int wid  = blockIdx.x * (blockDim.x >> 5) + (threadIdx.x >> 5);
  int lane = threadIdx.x & 31;
  int total = CE * HEADS;
  if (wid >= total) return;
  int e  = wid >> 2;
  int hd = wid & 3;
  int s, d;
  if (e < E) { s = src[e]; d = dst[e]; } else { s = d = e - E; }
  float a = expf(eedge[(long)e * HEADS + hd] - mmax[d * HEADS + hd]);
  if (lane == 0) atomicAdd(&denom[d * HEADS + hd], a);
  const float* hp = h + (long)s * OUT1 + hd * HID;
  float* ap = acc + (long)d * OUT1 + hd * HID;
  for (int c = lane; c < HID; c += 32)
    atomicAdd(&ap[c], hp[c] * a);
}

// out = relu(acc/(denom+eps) + bias), in-place on acc
__global__ void k_finalize_conv(float* __restrict__ acc, const float* __restrict__ denom,
                                const float* __restrict__ bias, long total) {
  long i = (long)blockIdx.x * blockDim.x + threadIdx.x;
  if (i >= total) return;
  int j  = (int)(i & (OUT1 - 1));
  long n = i >> 9;  // /OUT1
  int hd = j >> 7;
  float v = acc[i] / (denom[n * HEADS + hd] + EPS_DEN) + bias[j];
  acc[i] = v > 0.0f ? v : 0.0f;
}

// ---------------------------------------------------------------- label prop

__global__ void k_lp_res(const float* __restrict__ y, float* __restrict__ res, long n) {
  long i = (long)blockIdx.x * blockDim.x + threadIdx.x;
  if (i < n) res[i] = 0.5f * y[i];   // (1 - alpha) * y
}

// agg[dst,:] += out[src,:] * w. One wave per raw edge, 512 channels.
__global__ void k_lp_scatter(const int* __restrict__ src, const int* __restrict__ dst,
                             const float* __restrict__ ew,
                             const float* __restrict__ out, float* __restrict__ agg, int E) {
  int wid  = blockIdx.x * (blockDim.x >> 5) + (threadIdx.x >> 5);
  int lane = threadIdx.x & 31;
  if (wid >= E) return;
  int s = src[wid], d = dst[wid];
  float w = ew[wid];
  const float* op = out + (long)s * OUT1;
  float* ap = agg + (long)d * OUT1;
  for (int c = lane; c < OUT1; c += 32)
    atomicAdd(&ap[c], op[c] * w);
}

__global__ void k_lp_combine(const float* __restrict__ agg, const float* __restrict__ res,
                             float* __restrict__ out, long n) {
  long i = (long)blockIdx.x * blockDim.x + threadIdx.x;
  if (i >= n) return;
  float v = 0.5f * agg[i] + res[i];   // alpha * agg + res
  out[i] = fminf(fmaxf(v, 0.0f), 1.0f);
}

// ---------------------------------------------------------------- pooling + MLP

__global__ void k_pool(const float* __restrict__ x, const float* __restrict__ h1,
                       const float* __restrict__ h2, const int* __restrict__ batch,
                       float* __restrict__ pool, float* __restrict__ cnt, int N) {
  int n = blockIdx.x;
  if (n >= N) return;
  int b = batch[n];
  if (threadIdx.x == 0) atomicAdd(&cnt[b], 1.0f);
  for (int j = threadIdx.x; j < JK; j += blockDim.x) {
    float v;
    if (j < IN_CH)            v = x [(long)n * IN_CH + j];
    else if (j < IN_CH + OUT1) v = h1[(long)n * OUT1 + (j - IN_CH)];
    else                       v = h2[(long)n * OUT1 + (j - IN_CH - OUT1)];
    atomicAdd(&pool[(long)b * JK + j], v);
  }
}

__global__ void k_mlp1(const float* __restrict__ pool, const float* __restrict__ cnt,
                       const float* __restrict__ W, const float* __restrict__ bias,
                       float* __restrict__ out) {
  int g = blockIdx.x;       // 64 graphs
  int o = threadIdx.x;      // 256 outputs
  float inv = 1.0f / fmaxf(cnt[g], 1.0f);
  const float* pp = pool + (long)g * JK;
  float s = bias[o];
  for (int k = 0; k < JK; k++) s += pp[k] * inv * W[(long)k * MLP_H + o];
  out[g * MLP_H + o] = s > 0.0f ? s : 0.0f;
}

__global__ void k_mlp2(const float* __restrict__ hid, const float* __restrict__ W,
                       const float* __restrict__ bias, float* __restrict__ out) {
  int g = blockIdx.x;       // 64
  int o = threadIdx.x;      // 128
  const float* hp = hid + (long)g * MLP_H;
  float s = bias[o];
  for (int k = 0; k < MLP_H; k++) s += hp[k] * W[k * MLP_O + o];
  out[g * MLP_O + o] = s;
}

// ---------------------------------------------------------------- launch

extern "C" void kernel_launch(void* const* d_in, const int* in_sizes, int n_in,
                              void* d_out, int out_size, void* d_ws, size_t ws_size,
                              hipStream_t stream) {
  const float* x      = (const float*)d_in[0];
  const int*   ei     = (const int*)d_in[1];
  const int*   batch  = (const int*)d_in[2];
  const float* W1     = (const float*)d_in[3];
  const float* asrc1  = (const float*)d_in[4];
  const float* adst1  = (const float*)d_in[5];
  const float* bcv1   = (const float*)d_in[6];
  const float* W2     = (const float*)d_in[7];
  const float* asrc2  = (const float*)d_in[8];
  const float* adst2  = (const float*)d_in[9];
  const float* bcv2   = (const float*)d_in[10];
  const float* mW1    = (const float*)d_in[11];
  const float* mb1    = (const float*)d_in[12];
  const float* mW2    = (const float*)d_in[13];
  const float* mb2    = (const float*)d_in[14];

  const int N  = in_sizes[0] / IN_CH;
  const int E  = in_sizes[1] / 2;
  const int CE = E + N;
  const int* src = ei;
  const int* dst = ei + E;

  char* ws = (char*)d_ws;
  size_t off = 0;
  auto wsa = [&](size_t bytes) -> char* {
    char* p = ws + off;
    off += (bytes + 255) & ~(size_t)255;
    return p;
  };

  unsigned short* xb   = (unsigned short*)wsa((size_t)N * IN_CH * 2);
  unsigned short* w1b  = (unsigned short*)wsa((size_t)IN_CH * OUT1 * 2);
  unsigned short* w2b  = (unsigned short*)wsa((size_t)OUT1 * OUT1 * 2);
  unsigned short* h1b  = (unsigned short*)wsa((size_t)N * OUT1 * 2);
  float* bufA  = (float*)wsa((size_t)N * OUT1 * 4);   // GEMM out / LP agg
  float* bufB  = (float*)wsa((size_t)N * OUT1 * 4);   // conv1 acc -> h1 final
  float* bufC  = (float*)wsa((size_t)N * OUT1 * 4);   // conv2 acc -> h2 final
  float* bufD  = (float*)wsa((size_t)N * OUT1 * 4);   // LP residual
  float* es    = (float*)wsa((size_t)N * HEADS * 4);
  float* ed    = (float*)wsa((size_t)N * HEADS * 4);
  float* mmax  = (float*)wsa((size_t)N * HEADS * 4);
  float* denom = (float*)wsa((size_t)N * HEADS * 4);
  float* eedge = (float*)wsa((size_t)CE * HEADS * 4);
  float* deg   = (float*)wsa((size_t)N * 4);
  float* ew    = (float*)wsa((size_t)E * 4);
  float* pool  = (float*)wsa((size_t)NGRAPH * JK * 4);
  float* cnt   = (float*)wsa((size_t)NGRAPH * 4);
  float* hid1  = (float*)wsa((size_t)NGRAPH * MLP_H * 4);
  (void)ws_size; (void)n_in; (void)out_size;

  const long NF = (long)N * OUT1;
  auto fill = [&](float* p, float v, long n) {
    int blocks = (int)((n + 255) / 256);
    if (blocks > 65535) blocks = 65535;
    hipLaunchKernelGGL(k_fill, dim3(blocks), dim3(256), 0, stream, p, v, n);
  };

  // ---- graph normalization for label prop
  fill(deg, 0.0f, N);
  hipLaunchKernelGGL(k_deg,   dim3((E + 255) / 256), dim3(256), 0, stream, dst, deg, E);
  hipLaunchKernelGGL(k_edgew, dim3((E + 255) / 256), dim3(256), 0, stream, src, dst, deg, ew, E);

  // ---- bf16 conversions for GEMM operands
  hipLaunchKernelGGL(k_f32_to_bf16, dim3(4096), dim3(256), 0, stream, x,  xb,  (long)N * IN_CH);
  hipLaunchKernelGGL(k_f32_to_bf16, dim3(512),  dim3(256), 0, stream, W1, w1b, (long)IN_CH * OUT1);
  hipLaunchKernelGGL(k_f32_to_bf16, dim3(1024), dim3(256), 0, stream, W2, w2b, (long)OUT1 * OUT1);

  const int tilesM = (N + 15) / 16;
  const int tilesN = OUT1 / 16;
  const int totalTiles = tilesM * tilesN;
  const int gemmBlocks = (totalTiles + 3) / 4;        // 4 waves / 128-thread block

  const int logitWaves = N * HEADS;
  const int logitBlocks = (logitWaves + 3) / 4;
  const int aggWaves = CE * HEADS;
  const int aggBlocks = (aggWaves + 7) / 8;           // 8 waves / 256-thread block
  const int lpBlocks = (E + 7) / 8;

  auto run_conv = [&](const unsigned short* inBf, const unsigned short* wBf, int K,
                      const float* asrc, const float* adst, const float* bias,
                      float* acc) {
    hipLaunchKernelGGL(k_gemm_bf16, dim3(gemmBlocks), dim3(128), 0, stream,
                       (const __bf16*)inBf, (const __bf16*)wBf, bufA,
                       N, OUT1, K, tilesN, totalTiles);
    hipLaunchKernelGGL(k_attn_logits, dim3(logitBlocks), dim3(128), 0, stream,
                       bufA, asrc, adst, es, ed, N);
    fill(mmax, -INFINITY, (long)N * HEADS);
    hipLaunchKernelGGL(k_edge_logits_max, dim3((CE + 255) / 256), dim3(256), 0, stream,
                       src, dst, es, ed, eedge, mmax, E, CE);
    hipLaunchKernelGGL(k_fix_max, dim3((N * HEADS + 255) / 256), dim3(256), 0, stream,
                       mmax, (long)N * HEADS);
    fill(denom, 0.0f, (long)N * HEADS);
    fill(acc, 0.0f, NF);
    hipLaunchKernelGGL(k_edge_aggregate, dim3(aggBlocks), dim3(256), 0, stream,
                       src, dst, eedge, mmax, denom, bufA, acc, E, CE);
    hipLaunchKernelGGL(k_finalize_conv, dim3((int)((NF + 255) / 256)), dim3(256), 0, stream,
                       acc, denom, bias, NF);
  };

  auto run_lp = [&](float* y) {
    hipLaunchKernelGGL(k_lp_res, dim3((int)((NF + 255) / 256)), dim3(256), 0, stream,
                       y, bufD, NF);
    for (int it = 0; it < 2; it++) {
      fill(bufA, 0.0f, NF);
      hipLaunchKernelGGL(k_lp_scatter, dim3(lpBlocks), dim3(256), 0, stream,
                         src, dst, ew, y, bufA, E);
      hipLaunchKernelGGL(k_lp_combine, dim3((int)((NF + 255) / 256)), dim3(256), 0, stream,
                         bufA, bufD, y, NF);
    }
  };

  // ---- conv1 + LP  (h1 ends in bufB)
  run_conv(xb, w1b, IN_CH, asrc1, adst1, bcv1, bufB);
  run_lp(bufB);

  // ---- conv2 + LP  (h2 ends in bufC)
  hipLaunchKernelGGL(k_f32_to_bf16, dim3(8192), dim3(256), 0, stream, bufB, h1b, NF);
  run_conv(h1b, w2b, OUT1, asrc2, adst2, bcv2, bufC);
  run_lp(bufC);

  // ---- JK concat mean-pool + MLP head
  fill(pool, 0.0f, (long)NGRAPH * JK);
  fill(cnt, 0.0f, NGRAPH);
  hipLaunchKernelGGL(k_pool, dim3(N), dim3(256), 0, stream,
                     x, bufB, bufC, batch, pool, cnt, N);
  hipLaunchKernelGGL(k_mlp1, dim3(NGRAPH), dim3(MLP_H), 0, stream, pool, cnt, mW1, mb1, hid1);
  hipLaunchKernelGGL(k_mlp2, dim3(NGRAPH), dim3(MLP_O), 0, stream, hid1, mW2, mb2, (float*)d_out);
}
